// ImageCrossAttentionNet_32633161515426
// MI455X (gfx1250) — matrix operand order repaired and checked by reading, not compiled, not run on hardware
//
#include <hip/hip_runtime.h>

// ---------------------------------------------------------------------------
// Flash-attention-style fused branch kernel for the 9 (query-stream, key-
// stream) attention branches of ImageCrossAttentionNet on MI455X (gfx1250).
//
//  S = Q*K (bf16 WMMA, f32 acc) -> online softmax (row-sum via WMMA against
//  an all-ones fragment) -> O^T += P*V^T (WMMA) -> 1x1 conv as O^T * W^T
//  (WMMA) + bias + residual.
//
// V tiles are DMA'd into LDS by the Tensor Data Mover (double-buffered,
// TENSORcnt-tracked) so staging overlaps compute.
//
// wave32; block = 128 threads = 4 waves; each wave owns a 16-row tile.
// ---------------------------------------------------------------------------

typedef __bf16 bf16_t;
typedef __attribute__((ext_vector_type(16))) __bf16        v16bf;
typedef __attribute__((ext_vector_type(8)))  float         v8f;
typedef __attribute__((ext_vector_type(4)))  unsigned int  v4u;
typedef __attribute__((ext_vector_type(4)))  int           v4i;
typedef __attribute__((ext_vector_type(8)))  int           v8i;

#define N_TOK 4096      // W*H
#define CCH   64        // channels
#define BJ    64        // key/value column block
#define LDSS  72        // padded LDS row stride for bf16 tiles (elements)
#define VSTR  68        // padded LDS row stride for f32 V tiles (floats)

#if defined(__AMDGCN__) && __has_builtin(__builtin_amdgcn_tensor_load_to_lds) && \
    __has_builtin(__builtin_amdgcn_s_wait_tensorcnt)
#define HAS_TDM 1
#else
#define HAS_TDM 0
#endif

__device__ __forceinline__ v8f wmma_bf16(v16bf a, v16bf b, v8f c) {
  // D = A(16x32 bf16) * B(32x16 bf16) + C(16x16 f32)
  return __builtin_amdgcn_wmma_f32_16x16x32_bf16(
      /*neg_a=*/false, a, /*neg_b=*/false, b,
      /*c_mod=*/(short)0, c, /*reuse_a=*/false, /*reuse_b=*/false);
}

#if HAS_TDM
// TDM: DMA a 64x64 f32 tile (row stride 4096 elements) from global memory into
// LDS at lds_off, padding 4 DWORDs after every 64 DWORDs (-> 68-float rows).
// D# layout per CDNA5 ISA. This toolchain uses the
// 6-argument builtin form: (v4u g0, v8i g1, v4i g2, v4i g3, v8i extra, cpol).
__device__ __forceinline__ void tdm_load_tile_f32(unsigned long long ga,
                                                  unsigned lds_off) {
  const v4u g0 = {
      1u,                                               // count=1 (valid), user
      lds_off,                                          // lds_addr
      (unsigned)(ga & 0xffffffffu),                     // global_addr[31:0]
      (unsigned)((ga >> 32) & 0x1ffffffu) | (2u << 30)  // addr[56:32] | type=2
  };
  const v8i g1 = {
      (int)((2u << 16) |          // data_size = 4 bytes
            (1u << 20) |          // pad_enable
            (5u << 22) |          // pad_interval: 64 DWORDs
            (3u << 25)),          // pad_amount:   4 DWORDs
      (int)(4096u << 16),         // tensor_dim0[15:0] = 4096
      (int)(64u << 16),           // tensor_dim0[31:16]=0 | tensor_dim1 = 64
      (int)(64u << 16),           // tensor_dim1 hi = 0   | tile_dim0   = 64
      64,                         // tile_dim1 = 64, tile_dim2 = 0
      4096,                       // tensor_dim0_stride[31:0] = 4096
      0, 0                        // stride hi, tensor_dim1_stride = 0
  };
  const v4i gz4 = {0, 0, 0, 0};
  const v8i gz8 = {0, 0, 0, 0, 0, 0, 0, 0};
  __builtin_amdgcn_tensor_load_to_lds(g0, g1, gz4, gz4, gz8, 0);
}
#endif

__global__ __launch_bounds__(128)
void flash_branch_kernel(const float* __restrict__ Qg,   // [B,N,C]
                         const float* __restrict__ Kg,   // [B,C,N]
                         const float* __restrict__ Vg,   // [B,C,N]
                         const float* __restrict__ Xg,   // [B,C,N] residual
                         const float* __restrict__ Wg,   // [C,C] 1x1 conv
                         const float* __restrict__ Bias, // [C]
                         float*       __restrict__ Outg) // -> [B,3C,N] slice
{
  __shared__ bf16_t Kt[BJ][LDSS];        // K^T tile: Kt[jj][c]  (bf16)
  __shared__ float  Vf[2][CCH][VSTR];    // V tiles:  Vf[buf][c][jj] (f32, TDM)
  __shared__ bf16_t Pb[4][16][LDSS];     // per-wave D->A relayout staging

  const int tid  = threadIdx.x;
  const int wave = tid >> 5;
  const int lane = tid & 31;
  const int col  = lane & 15;            // N-col of B/C/D frags, row of A frag
  const int hi   = lane >> 4;            // lane half selects K range / row+8
  const int b    = blockIdx.y;

  const float* Q   = Qg + (size_t)b * N_TOK * CCH;
  const float* K   = Kg + (size_t)b * CCH * N_TOK;
  const float* V   = Vg + (size_t)b * CCH * N_TOK;
  const float* X   = Xg + (size_t)b * CCH * N_TOK;
  float*       Out = Outg + (size_t)b * 3 * CCH * N_TOK;

  const int rowTile = blockIdx.x * 64 + wave * 16;   // first query row

  // ---- Q A-fragments: rows rowTile..+15, K = 64 channels (2 chunks of 32) --
  // A 16x32 layout: lane row = col; elem e<8 -> k = 32*ch + 8*hi + e,
  //                               e>=8 -> k = 32*ch + 8*hi + 16 + (e-8)
  v16bf qa[2];
  {
    const float* qrow = Q + (size_t)(rowTile + col) * CCH;
#pragma unroll
    for (int ch = 0; ch < 2; ++ch) {
      const int base = ch * 32 + hi * 8;
#pragma unroll
      for (int e = 0; e < 8; ++e) {
        qa[ch][e]     = (bf16_t)qrow[base + e];
        qa[ch][e + 8] = (bf16_t)qrow[base + 16 + e];
      }
    }
  }

  // All-ones B-fragment: row sums of P via the matrix unit.
  v16bf onesb;
#pragma unroll
  for (int e = 0; e < 16; ++e) onesb[e] = (bf16_t)1.0f;

  // Online-softmax state. D-slot r <-> query row (rowTile + r + 8*hi).
  float m_r[8], l_r[8];
  v8f   oacc[4];                         // O^T accum: tile nt = channels 16nt..
  const v8f vzero = {0.f, 0.f, 0.f, 0.f, 0.f, 0.f, 0.f, 0.f};
#pragma unroll
  for (int r = 0; r < 8; ++r) { m_r[r] = -3.0e38f; l_r[r] = 0.0f; }
#pragma unroll
  for (int nt = 0; nt < 4; ++nt) oacc[nt] = vzero;

#if HAS_TDM
  // Prologue: DMA first V tile into buffer 0 while we do nothing else yet.
  if (wave == 0)
    tdm_load_tile_f32((unsigned long long)(const void*)V,
                      (unsigned)(size_t)(&Vf[0][0][0]));
#endif

  for (int jb = 0, it = 0; jb < N_TOK; jb += BJ, ++it) {
    const int cur = it & 1;
    __syncthreads();     // previous block's reads of Kt / Vf[cur^1] are done

    // ---- stage K^T block into LDS as bf16 (coalesced over jj) -------------
    for (int idx = tid; idx < CCH * BJ; idx += 128) {
      const int c  = idx >> 6;
      const int jj = idx & 63;
      Kt[jj][c] = (bf16_t)K[(size_t)c * N_TOK + jb + jj];
    }
    if (jb + BJ < N_TOK) {   // pull next K block toward L2 (global_prefetch)
      const size_t pg = (size_t)(tid >> 1) * N_TOK + jb + BJ + (size_t)(tid & 1) * 32;
      __builtin_prefetch(&K[pg], 0, 1);
    }

#if HAS_TDM
    // ---- V tiles via Tensor Data Mover, double-buffered -------------------
    if (wave == 0) {
      if (jb + BJ < N_TOK) {
        tdm_load_tile_f32((unsigned long long)(const void*)(V + jb + BJ),
                          (unsigned)(size_t)(&Vf[cur ^ 1][0][0]));
        __builtin_amdgcn_s_wait_tensorcnt(1);   // current buffer's DMA done
      } else {
        __builtin_amdgcn_s_wait_tensorcnt(0);
      }
    }
#else
    for (int idx = tid; idx < CCH * BJ; idx += 128) {
      const int c  = idx >> 6;
      const int jj = idx & 63;
      Vf[cur][c][jj] = V[(size_t)c * N_TOK + jb + jj];
    }
#endif
    __syncthreads();     // Kt staged + Vf[cur] DMA complete, visible to all

    // ---- S(16x64) = Q_tile * K_block : 8 WMMAs ----------------------------
    // B 32x16 layout: lane col = jj; elem e -> k = 32*ch + 8*hi + (e&7) + 16*(e>>3)
    v8f s[4];
#pragma unroll
    for (int nt = 0; nt < 4; ++nt) {
      const bf16_t* kp = &Kt[nt * 16 + col][0];
      v16bf kb0, kb1;
#pragma unroll
      for (int e = 0; e < 16; ++e) {
        kb0[e] = kp[hi * 8 + (e & 7) + (e >> 3) * 16];
        kb1[e] = kp[32 + hi * 8 + (e & 7) + (e >> 3) * 16];
      }
      v8f acc = vzero;
      acc = wmma_bf16(qa[0], kb0, acc);
      acc = wmma_bf16(qa[1], kb1, acc);
      s[nt] = acc;
    }

    // ---- online softmax: row max across 16 lanes of each half -------------
    float corr[8];
#pragma unroll
    for (int r = 0; r < 8; ++r) {
      float mx = fmaxf(fmaxf(s[0][r], s[1][r]), fmaxf(s[2][r], s[3][r]));
#pragma unroll
      for (int off = 1; off < 16; off <<= 1)
        mx = fmaxf(mx, __shfl_xor(mx, off, 32));
      const float mnew = fmaxf(m_r[r], mx);
      corr[r] = __expf(m_r[r] - mnew);
      m_r[r]  = mnew;
#pragma unroll
      for (int nt = 0; nt < 4; ++nt) {
        s[nt][r] = __expf(s[nt][r] - mnew);
        oacc[nt][r] *= corr[r];
      }
    }

    // ---- relayout P: D layout -> A layout via per-wave LDS staging --------
#pragma unroll
    for (int nt = 0; nt < 4; ++nt)
#pragma unroll
      for (int r = 0; r < 8; ++r)
        Pb[wave][r + 8 * hi][nt * 16 + col] = (bf16_t)s[nt][r];

    v16bf pa[2];
    {
      const bf16_t* prow = &Pb[wave][col][0];
#pragma unroll
      for (int ch = 0; ch < 2; ++ch) {
        const int base = ch * 32 + hi * 8;
#pragma unroll
        for (int e = 0; e < 8; ++e) {
          pa[ch][e]     = prow[base + e];
          pa[ch][e + 8] = prow[base + 16 + e];
        }
      }
    }

    // ---- row sums of P on the matrix unit: P @ ones -> D-slot r = row sum -
    {
      v8f ls = vzero;
      ls = wmma_bf16(pa[0], onesb, ls);
      ls = wmma_bf16(pa[1], onesb, ls);
#pragma unroll
      for (int r = 0; r < 8; ++r) l_r[r] = l_r[r] * corr[r] + ls[r];
    }

    // ---- O^T(16x64) += P(16x64) * V^T(64x64) : 8 WMMAs --------------------
    // B-frag from f32 V tile: lane col -> channel, elem e -> j = 32ch+8hi+(e&7)+16(e>>3)
#pragma unroll
    for (int nt = 0; nt < 4; ++nt) {
      const float* vp = &Vf[cur][nt * 16 + col][0];
      v16bf vb0, vb1;
#pragma unroll
      for (int e = 0; e < 16; ++e) {
        vb0[e] = (bf16_t)vp[hi * 8 + (e & 7) + (e >> 3) * 16];
        vb1[e] = (bf16_t)vp[32 + hi * 8 + (e & 7) + (e >> 3) * 16];
      }
      oacc[nt] = wmma_bf16(pa[0], vb0, oacc[nt]);
      oacc[nt] = wmma_bf16(pa[1], vb1, oacc[nt]);
    }
  }

  // ---- normalize and relayout O^T for the 1x1-conv GEMM ---------------------
#pragma unroll
  for (int r = 0; r < 8; ++r) {
    const float inv = 1.0f / l_r[r];
#pragma unroll
    for (int nt = 0; nt < 4; ++nt)
      Pb[wave][r + 8 * hi][nt * 16 + col] = (bf16_t)(oacc[nt][r] * inv);
  }

  v16bf oa[2];
  {
    const bf16_t* prow = &Pb[wave][col][0];
#pragma unroll
    for (int ch = 0; ch < 2; ++ch) {
      const int base = ch * 32 + hi * 8;
#pragma unroll
      for (int e = 0; e < 8; ++e) {
        oa[ch][e]     = prow[base + e];
        oa[ch][e + 8] = prow[base + 16 + e];
      }
    }
  }

  // ---- out^T(16x64) = O^T * W^T : 8 WMMAs, then bias + residual + store ----
#pragma unroll
  for (int nt = 0; nt < 4; ++nt) {
    const int o = nt * 16 + col;
    const float* wrow = Wg + (size_t)o * CCH;   // B[k=c][n=o] = W[o][c]
    v16bf wb0, wb1;
#pragma unroll
    for (int e = 0; e < 16; ++e) {
      wb0[e] = (bf16_t)wrow[hi * 8 + (e & 7) + (e >> 3) * 16];
      wb1[e] = (bf16_t)wrow[32 + hi * 8 + (e & 7) + (e >> 3) * 16];
    }
    v8f d = vzero;
    d = wmma_bf16(oa[0], wb0, d);
    d = wmma_bf16(oa[1], wb1, d);
    const float bo = Bias[o];
#pragma unroll
    for (int r = 0; r < 8; ++r) {
      const int i = rowTile + r + 8 * hi;
      const size_t g = (size_t)o * N_TOK + i;
      Out[g] = d[r] + bo + X[g];
    }
  }
}

// ---------------------------------------------------------------------------
// Host launcher: 9 branch launches (3 query streams x 3 key streams).
// Input dict order: [qs*4 + 0]=q, [+1]=k, [+2]=v, [+3]=x for qs in
// {poi,point,pop}; then w_poi,b_poi,w_point,b_point,w_pop,b_pop (12..17).
// Output: concat of 3 tensors [B, 3C, W, H]; branch (qs,ks,b) slice starts at
// qs*B*3C*N + b*3C*N + ks*C*N.
// ---------------------------------------------------------------------------
extern "C" void kernel_launch(void* const* d_in, const int* in_sizes, int n_in,
                              void* d_out, int out_size, void* d_ws, size_t ws_size,
                              hipStream_t stream) {
  (void)in_sizes; (void)n_in; (void)out_size; (void)d_ws; (void)ws_size;
  const int Bb = 2, C = CCH, N = N_TOK;

  for (int qs = 0; qs < 3; ++qs) {
    const float* Q   = (const float*)d_in[4 * qs + 0];
    const float* V   = (const float*)d_in[4 * qs + 2];
    const float* X   = (const float*)d_in[4 * qs + 3];
    const float* Wc  = (const float*)d_in[12 + 2 * qs];
    const float* Bi  = (const float*)d_in[13 + 2 * qs];
    for (int ks = 0; ks < 3; ++ks) {
      const float* K = (const float*)d_in[4 * ks + 1];
      float* Out = (float*)d_out + (size_t)qs * Bb * 3 * C * N + (size_t)ks * C * N;
      dim3 grid(N / 64, Bb);
      flash_branch_kernel<<<grid, dim3(128), 0, stream>>>(Q, K, V, X, Wc, Bi, Out);
    }
  }
}
